// CerebroOriginalBlock_79182017069527
// MI455X (gfx1250) — 16-bit and 8-bit matrix operand order repaired and checked by reading, compiled, not run on hardware
//
#include <hip/hip_runtime.h>
#include <hip/hip_fp16.h>
#include <cmath>
#include <cstdint>


typedef __attribute__((ext_vector_type(4)))  int      v4i;
typedef __attribute__((ext_vector_type(8)))  int      v8i;
typedef __attribute__((ext_vector_type(8)))  float    v8f;
typedef __attribute__((ext_vector_type(16))) _Float16 v16h;

#define DMODEL 768
#define MROWS  32768
#define NBATCH 8
#define CCH    64
#define TLEN   64
#define NHEAD  12
#define DHEAD  64
#define QMAXF  127.0f

union AFrag8 { v8i  v; int2 d[4]; };
union BFrag8 { v8i  v; int4 q[2]; };
union HFrag  { v16h v; int4 q[2]; };

#if defined(__gfx1250__) &&                                                   \
    __has_builtin(__builtin_amdgcn_global_load_async_to_lds_b128) &&          \
    __has_builtin(__builtin_amdgcn_s_wait_asynccnt)
#define USE_ASYNC_LDS 1
static __device__ __forceinline__ void async_copy_b128(void* lds_dst,
                                                       const void* gsrc) {
  __builtin_amdgcn_global_load_async_to_lds_b128(
      (__attribute__((address_space(1))) v4i*)gsrc,
      (__attribute__((address_space(3))) v4i*)lds_dst, 0, 0);
}
#else
#define USE_ASYNC_LDS 0
#endif

static __device__ __forceinline__ float warp_max32(float v) {
#pragma unroll
  for (int m = 16; m >= 1; m >>= 1) v = fmaxf(v, __shfl_xor(v, m, 32));
  return v;
}
static __device__ __forceinline__ float warp_sum32(float v) {
#pragma unroll
  for (int m = 16; m >= 1; m >>= 1) v += __shfl_xor(v, m, 32);
  return v;
}

__global__ void init_amax_kernel(unsigned int* amax) {
  if (threadIdx.x < 8) amax[threadIdx.x] = 0u;
}

__global__ void absmax_f32_kernel(const float* __restrict__ x, int n,
                                  unsigned int* __restrict__ slot) {
  float m = 0.f;
  for (int i = blockIdx.x * blockDim.x + threadIdx.x; i < n;
       i += gridDim.x * blockDim.x)
    m = fmaxf(m, fabsf(x[i]));
  m = warp_max32(m);
  if ((threadIdx.x & 31) == 0) atomicMax(slot, __float_as_uint(m));
}

__global__ void quant_f32_kernel(const float* __restrict__ x, int n,
                                 const unsigned int* __restrict__ slot,
                                 int8_t* __restrict__ q) {
  const float s = QMAXF / fmaxf(__uint_as_float(*slot), 1e-8f);
  for (int i = blockIdx.x * blockDim.x + threadIdx.x; i < n;
       i += gridDim.x * blockDim.x) {
    float v = rintf(x[i] * s);
    v = fminf(fmaxf(v, -128.f), 127.f);
    q[i] = (int8_t)v;
  }
}

__global__ void quant_f16_kernel(const _Float16* __restrict__ x, int n,
                                 const unsigned int* __restrict__ slot,
                                 int8_t* __restrict__ q) {
  const float s = QMAXF / fmaxf(__uint_as_float(*slot), 1e-8f);
  for (int i = blockIdx.x * blockDim.x + threadIdx.x; i < n;
       i += gridDim.x * blockDim.x) {
    float v = rintf((float)x[i] * s);
    v = fminf(fmaxf(v, -128.f), 127.f);
    q[i] = (int8_t)v;
  }
}

__global__ __launch_bounds__(256)
void ln_stats_absmax_kernel(const float* __restrict__ x,
                            const float* __restrict__ g,
                            const float* __restrict__ b,
                            float* __restrict__ mean, float* __restrict__ rstd,
                            unsigned int* __restrict__ slot) {
  const int row  = blockIdx.x * 8 + (threadIdx.x >> 5);
  const int lane = threadIdx.x & 31;
  const float* xr = x + (size_t)row * DMODEL;
  float s = 0.f, ss = 0.f;
  for (int i = lane; i < DMODEL; i += 32) { float v = xr[i]; s += v; ss += v * v; }
  s = warp_sum32(s); ss = warp_sum32(ss);
  const float m   = s * (1.0f / DMODEL);
  const float var = ss * (1.0f / DMODEL) - m * m;
  const float rs  = rsqrtf(var + 1e-5f);
  if (lane == 0) { mean[row] = m; rstd[row] = rs; }
  float am = 0.f;
  for (int i = lane; i < DMODEL; i += 32) {
    float h = (xr[i] - m) * rs * g[i] + b[i];
    am = fmaxf(am, fabsf(h));
  }
  am = warp_max32(am);
  if (lane == 0) atomicMax(slot, __float_as_uint(am));
}

__global__ __launch_bounds__(256)
void ln_quant_kernel(const float* __restrict__ x,
                     const float* __restrict__ g, const float* __restrict__ b,
                     const float* __restrict__ mean,
                     const float* __restrict__ rstd,
                     const unsigned int* __restrict__ slot,
                     int8_t* __restrict__ q) {
  const int row  = blockIdx.x * 8 + (threadIdx.x >> 5);
  const int lane = threadIdx.x & 31;
  const float s  = QMAXF / fmaxf(__uint_as_float(*slot), 1e-8f);
  const float m  = mean[row], rs = rstd[row];
  const float* xr = x + (size_t)row * DMODEL;
  int8_t* qr = q + (size_t)row * DMODEL;
  for (int i = lane; i < DMODEL; i += 32) {
    float h = ((xr[i] - m) * rs * g[i] + b[i]) * s;
    h = fminf(fmaxf(rintf(h), -128.f), 127.f);
    qr[i] = (int8_t)h;
  }
}

template <int EPI>
__global__ __launch_bounds__(256)
void gemm_i8_kernel(const int8_t* __restrict__ A, const int8_t* __restrict__ Bw,
                    int M, int N, int K,
                    const unsigned int* __restrict__ amaxA,
                    const unsigned int* __restrict__ amaxB,
                    const float* __restrict__ bias,
                    const float* __restrict__ resid,
                    float* __restrict__ out32, _Float16* __restrict__ out16,
                    unsigned int* __restrict__ amax_out) {
  __shared__ int8_t As[2][128][80];
  __shared__ int8_t Bs[2][256][80];

  const int t    = threadIdx.x;
  const int lane = t & 31;
  const int wid  = t >> 5;
  const int wm   = wid & 1;
  const int wn   = wid >> 1;
  const int hi   = lane >> 4;
  const int nl   = lane & 15;
  const int m0   = blockIdx.y * 128;
  const int n0   = blockIdx.x * 256;

  const int arow = t >> 1;
  const int aoff = (t & 1) * 32;

  v8i acc[4][4] = {};

#if USE_ASYNC_LDS
  {
    const int8_t* ga = A + (size_t)(m0 + arow) * K + aoff;
    async_copy_b128(&As[0][arow][aoff],      ga);
    async_copy_b128(&As[0][arow][aoff + 16], ga + 16);
    const int8_t* gb = Bw + (size_t)(n0 + t) * K;
#pragma unroll
    for (int j = 0; j < 4; ++j)
      async_copy_b128(&Bs[0][t][j * 16], gb + j * 16);
  }
#endif

  for (int k0 = 0; k0 < K; k0 += 64) {
    const int buf = (k0 >> 6) & 1;
#if USE_ASYNC_LDS
    __builtin_amdgcn_s_wait_asynccnt(0);
    __syncthreads();
    if (k0 + 64 < K) {
      const int nb = buf ^ 1;
      const int8_t* ga = A + (size_t)(m0 + arow) * K + (k0 + 64) + aoff;
      async_copy_b128(&As[nb][arow][aoff],      ga);
      async_copy_b128(&As[nb][arow][aoff + 16], ga + 16);
      const int8_t* gb = Bw + (size_t)(n0 + t) * K + (k0 + 64);
#pragma unroll
      for (int j = 0; j < 4; ++j)
        async_copy_b128(&Bs[nb][t][j * 16], gb + j * 16);
    }
#else
    {
      const int4* ga = (const int4*)(A + (size_t)(m0 + arow) * K + k0 + aoff);
      int4* da = (int4*)&As[buf][arow][aoff];
      da[0] = ga[0]; da[1] = ga[1];
      const int4* gb = (const int4*)(Bw + (size_t)(n0 + t) * K + k0);
      int4* db = (int4*)&Bs[buf][t][0];
#pragma unroll
      for (int j = 0; j < 4; ++j) db[j] = gb[j];
    }
    __syncthreads();
#endif

    AFrag8 af[4];
#pragma unroll
    for (int mt = 0; mt < 4; ++mt) {
      const int8_t* src = &As[buf][wm * 64 + mt * 16 + nl][hi * 16];
#pragma unroll
      for (int p = 0; p < 4; ++p) af[mt].d[p] = *(const int2*)(src + (p >> 1) * 32 + (p & 1) * 8);
    }
    BFrag8 bf[4];
#pragma unroll
    for (int nt = 0; nt < 4; ++nt) {
      const int8_t* src = &Bs[buf][wn * 64 + nt * 16 + nl][hi * 16];
      bf[nt].q[0] = *(const int4*)(src);
      bf[nt].q[1] = *(const int4*)(src + 32);
    }
#pragma unroll
    for (int mt = 0; mt < 4; ++mt)
#pragma unroll
      for (int nt = 0; nt < 4; ++nt)
        acc[mt][nt] = __builtin_amdgcn_wmma_i32_16x16x64_iu8(
            true, af[mt].v, true, bf[nt].v, acc[mt][nt], false, false);
#if !USE_ASYNC_LDS
    __syncthreads();
#endif
  }

  const float sa = fmaxf(__uint_as_float(*amaxA), 1e-8f) * (1.0f / QMAXF);
  const float sb = fmaxf(__uint_as_float(*amaxB), 1e-8f) * (1.0f / QMAXF);
  const float ds = sa * sb;
  float lamax = 0.f;

#pragma unroll
  for (int mt = 0; mt < 4; ++mt) {
#pragma unroll
    for (int nt = 0; nt < 4; ++nt) {
      const int n = n0 + wn * 64 + nt * 16 + nl;
      const float bn = bias[n];
#pragma unroll
      for (int r = 0; r < 8; ++r) {
        const int m = m0 + wm * 64 + mt * 16 + hi * 8 + r;
        const size_t idx = (size_t)m * N + n;
        float f = (float)acc[mt][nt][r] * ds + bn;
        if (EPI == 0) {
          out16[idx] = (_Float16)f;
        } else if (EPI == 1) {
          out32[idx] = f + resid[idx];
        } else {
          float gl = 0.5f * f * (1.0f + erff(f * 0.70710678118654752f));
          out16[idx] = (_Float16)gl;
          lamax = fmaxf(lamax, fabsf(gl));
        }
      }
    }
  }
  if (EPI == 2) {
    lamax = warp_max32(lamax);
    if (lane == 0) atomicMax(amax_out, __float_as_uint(lamax));
  }
}

__global__ __launch_bounds__(128)
void attn_kernel(const _Float16* __restrict__ qkv, _Float16* __restrict__ o,
                 unsigned int* __restrict__ amax_o) {
  __shared__ _Float16 qs[64][72];
  __shared__ _Float16 ks[64][72];
  __shared__ _Float16 vt[64][72];
  __shared__ _Float16 ps[64][72];

  const int g   = blockIdx.x;
  const int b   = g / (CCH * NHEAD);
  const int rem = g % (CCH * NHEAD);
  const int c   = rem / NHEAD;
  const int h   = rem % NHEAD;

  const int t    = threadIdx.x;
  const int lane = t & 31;
  const int w    = t >> 5;
  const int hi   = lane >> 4;
  const int nl   = lane & 15;

  const size_t rowbase = (size_t)b * (CCH * TLEN) + (size_t)c * TLEN;
  const int cq = h * DHEAD;
  const int ck = DMODEL + h * DHEAD;
  const int cv = 2 * DMODEL + h * DHEAD;

#pragma unroll
  for (int j = 0; j < 4; ++j) {
    const int e  = t + j * 128;
    const int tt = e >> 3;
    const int dd = (e & 7) * 8;
    const _Float16* base = qkv + (rowbase + tt) * (3 * DMODEL);
    *(int4*)&qs[tt][dd] = *(const int4*)(base + cq + dd);
    *(int4*)&ks[tt][dd] = *(const int4*)(base + ck + dd);
    int4 vv = *(const int4*)(base + cv + dd);
    const _Float16* vh = (const _Float16*)&vv;
#pragma unroll
    for (int i = 0; i < 8; ++i) vt[dd + i][tt] = vh[i];
  }
  __syncthreads();

  const int t0 = w * 16;

  v8f sacc[4] = {};
#pragma unroll
  for (int kk = 0; kk < 2; ++kk) {
    HFrag aq;
    const _Float16* arow = &qs[t0 + nl][kk * 32 + hi * 16];
    aq.q[0] = *(const int4*)(arow);
    aq.q[1] = *(const int4*)(arow + 8);
#pragma unroll
    for (int nt = 0; nt < 4; ++nt) {
      HFrag bk;
      const _Float16* brow = &ks[nt * 16 + nl][kk * 32 + hi * 16];
      bk.q[0] = *(const int4*)(brow);
      bk.q[1] = *(const int4*)(brow + 8);
      sacc[nt] = __builtin_amdgcn_wmma_f32_16x16x32_f16(
          false, aq.v, false, bk.v, (short)0, sacc[nt], false, false);
    }
  }

  const float scl = 0.125f;
#pragma unroll
  for (int r = 0; r < 8; ++r) {
    float v[4];
    float rmax = -3.4e38f;
#pragma unroll
    for (int nt = 0; nt < 4; ++nt) {
      v[nt] = sacc[nt][r] * scl;
      rmax = fmaxf(rmax, v[nt]);
    }
#pragma unroll
    for (int m = 8; m >= 1; m >>= 1) rmax = fmaxf(rmax, __shfl_xor(rmax, m, 32));
    float rsum = 0.f;
#pragma unroll
    for (int nt = 0; nt < 4; ++nt) { v[nt] = __expf(v[nt] - rmax); rsum += v[nt]; }
#pragma unroll
    for (int m = 8; m >= 1; m >>= 1) rsum += __shfl_xor(rsum, m, 32);
    const float inv = 1.0f / rsum;
    const int row = t0 + hi * 8 + r;
#pragma unroll
    for (int nt = 0; nt < 4; ++nt) ps[row][nt * 16 + nl] = (_Float16)(v[nt] * inv);
  }
  __syncthreads();

  v8f oacc[4] = {};
#pragma unroll
  for (int kk = 0; kk < 2; ++kk) {
    HFrag ap;
    const _Float16* arow = &ps[t0 + nl][kk * 32 + hi * 16];
    ap.q[0] = *(const int4*)(arow);
    ap.q[1] = *(const int4*)(arow + 8);
#pragma unroll
    for (int nd = 0; nd < 4; ++nd) {
      HFrag bv;
      const _Float16* brow = &vt[nd * 16 + nl][kk * 32 + hi * 16];
      bv.q[0] = *(const int4*)(brow);
      bv.q[1] = *(const int4*)(brow + 8);
      oacc[nd] = __builtin_amdgcn_wmma_f32_16x16x32_f16(
          false, ap.v, false, bv.v, (short)0, oacc[nd], false, false);
    }
  }

  float lamax = 0.f;
#pragma unroll
  for (int nd = 0; nd < 4; ++nd) {
#pragma unroll
    for (int r = 0; r < 8; ++r) {
      const float f = oacc[nd][r];
      const size_t idx =
          (rowbase + t0 + hi * 8 + r) * DMODEL + h * DHEAD + nd * 16 + nl;
      o[idx] = (_Float16)f;
      lamax = fmaxf(lamax, fabsf(f));
    }
  }
  lamax = warp_max32(lamax);
  if (lane == 0) atomicMax(amax_o, __float_as_uint(lamax));
}

extern "C" void kernel_launch(void* const* d_in, const int* in_sizes, int n_in,
                              void* d_out, int out_size, void* d_ws,
                              size_t ws_size, hipStream_t stream) {
  (void)in_sizes; (void)n_in; (void)out_size; (void)ws_size;
  const float* x      = (const float*)d_in[0];
  const float* ln1_g  = (const float*)d_in[1];
  const float* ln1_b  = (const float*)d_in[2];
  const float* qkv_w  = (const float*)d_in[3];
  const float* qkv_b  = (const float*)d_in[4];
  const float* proj_w = (const float*)d_in[5];
  const float* proj_b = (const float*)d_in[6];
  const float* ln2_g  = (const float*)d_in[7];
  const float* ln2_b  = (const float*)d_in[8];
  const float* fc1_w  = (const float*)d_in[9];
  const float* fc1_b  = (const float*)d_in[10];
  const float* fc2_w  = (const float*)d_in[11];
  const float* fc2_b  = (const float*)d_in[12];
  float* out = (float*)d_out;

  char* ws = (char*)d_ws;
  size_t off = 0;
  auto take = [&](size_t bytes) -> char* {
    char* p = ws + off;
    off += (bytes + 255) & ~(size_t)255;
    return p;
  };

  unsigned int* amax = (unsigned int*)take(64);
  float*    mean1 = (float*)take((size_t)MROWS * 4);
  float*    rstd1 = (float*)take((size_t)MROWS * 4);
  int8_t*   qkvwq = (int8_t*)take((size_t)3 * DMODEL * DMODEL);
  int8_t*   projwq= (int8_t*)take((size_t)DMODEL * DMODEL);
  int8_t*   fc1wq = (int8_t*)take((size_t)4 * DMODEL * DMODEL);
  int8_t*   fc2wq = (int8_t*)take((size_t)4 * DMODEL * DMODEL);
  int8_t*   h1q   = (int8_t*)take((size_t)MROWS * DMODEL);
  _Float16* qkvh  = (_Float16*)take((size_t)MROWS * 3 * DMODEL * 2);
  _Float16* oh    = (_Float16*)take((size_t)MROWS * DMODEL * 2);
  int8_t*   oq    = (int8_t*)take((size_t)MROWS * DMODEL);
  float*    x2    = (float*)take((size_t)MROWS * DMODEL * 4);
  int8_t*   h2q   = (int8_t*)take((size_t)MROWS * DMODEL);
  _Float16* gh    = (_Float16*)take((size_t)MROWS * 4 * DMODEL * 2);
  int8_t*   gq    = (int8_t*)take((size_t)MROWS * 4 * DMODEL);

  hipLaunchKernelGGL(init_amax_kernel, dim3(1), dim3(32), 0, stream, amax);

  const int n_qkvw = 3 * DMODEL * DMODEL, n_projw = DMODEL * DMODEL,
            n_fcw  = 4 * DMODEL * DMODEL;
  hipLaunchKernelGGL(absmax_f32_kernel, dim3(512), dim3(256), 0, stream, qkv_w,  n_qkvw, amax + 0);
  hipLaunchKernelGGL(absmax_f32_kernel, dim3(512), dim3(256), 0, stream, proj_w, n_projw, amax + 1);
  hipLaunchKernelGGL(absmax_f32_kernel, dim3(512), dim3(256), 0, stream, fc1_w,  n_fcw,  amax + 2);
  hipLaunchKernelGGL(absmax_f32_kernel, dim3(512), dim3(256), 0, stream, fc2_w,  n_fcw,  amax + 3);
  hipLaunchKernelGGL(quant_f32_kernel, dim3(2048), dim3(256), 0, stream, qkv_w,  n_qkvw, amax + 0, qkvwq);
  hipLaunchKernelGGL(quant_f32_kernel, dim3(2048), dim3(256), 0, stream, proj_w, n_projw, amax + 1, projwq);
  hipLaunchKernelGGL(quant_f32_kernel, dim3(2048), dim3(256), 0, stream, fc1_w,  n_fcw,  amax + 2, fc1wq);
  hipLaunchKernelGGL(quant_f32_kernel, dim3(2048), dim3(256), 0, stream, fc2_w,  n_fcw,  amax + 3, fc2wq);

  hipLaunchKernelGGL(ln_stats_absmax_kernel, dim3(MROWS / 8), dim3(256), 0, stream,
                     x, ln1_g, ln1_b, mean1, rstd1, amax + 4);
  hipLaunchKernelGGL(ln_quant_kernel, dim3(MROWS / 8), dim3(256), 0, stream,
                     x, ln1_g, ln1_b, mean1, rstd1, amax + 4, h1q);

  hipLaunchKernelGGL((gemm_i8_kernel<0>), dim3(3 * DMODEL / 256, MROWS / 128),
                     dim3(256), 0, stream, h1q, qkvwq, MROWS, 3 * DMODEL, DMODEL,
                     amax + 4, amax + 0, qkv_b, (const float*)nullptr,
                     (float*)nullptr, qkvh, (unsigned int*)nullptr);

  hipLaunchKernelGGL(attn_kernel, dim3(NBATCH * CCH * NHEAD), dim3(128), 0,
                     stream, qkvh, oh, amax + 5);
  hipLaunchKernelGGL(quant_f16_kernel, dim3(4096), dim3(256), 0, stream, oh,
                     MROWS * DMODEL, amax + 5, oq);

  hipLaunchKernelGGL((gemm_i8_kernel<1>), dim3(DMODEL / 256, MROWS / 128),
                     dim3(256), 0, stream, oq, projwq, MROWS, DMODEL, DMODEL,
                     amax + 5, amax + 1, proj_b, x, x2, (_Float16*)nullptr,
                     (unsigned int*)nullptr);

  hipLaunchKernelGGL(ln_stats_absmax_kernel, dim3(MROWS / 8), dim3(256), 0, stream,
                     x2, ln2_g, ln2_b, mean1, rstd1, amax + 6);
  hipLaunchKernelGGL(ln_quant_kernel, dim3(MROWS / 8), dim3(256), 0, stream,
                     x2, ln2_g, ln2_b, mean1, rstd1, amax + 6, h2q);

  hipLaunchKernelGGL((gemm_i8_kernel<2>), dim3(4 * DMODEL / 256, MROWS / 128),
                     dim3(256), 0, stream, h2q, fc1wq, MROWS, 4 * DMODEL, DMODEL,
                     amax + 6, amax + 2, fc1_b, (const float*)nullptr,
                     (float*)nullptr, gh, amax + 7);
  hipLaunchKernelGGL(quant_f16_kernel, dim3(8192), dim3(256), 0, stream, gh,
                     MROWS * 4 * DMODEL, amax + 7, gq);

  hipLaunchKernelGGL((gemm_i8_kernel<1>), dim3(DMODEL / 256, MROWS / 128),
                     dim3(256), 0, stream, gq, fc2wq, MROWS, DMODEL, 4 * DMODEL,
                     amax + 7, amax + 3, fc2_b, x2, out, (_Float16*)nullptr,
                     (unsigned int*)nullptr);
}
